// CBiAFormerBlock_88751204205353
// MI455X (gfx1250) — matrix operand order repaired and checked by reading, not compiled, not run on hardware
//
#include <hip/hip_runtime.h>
#include <hip/hip_bf16.h>

// ---------------------------------------------------------------------------
// Types for CDNA5 WMMA (wave32): v_wmma_f32_16x16x32_bf16
// ---------------------------------------------------------------------------
typedef __attribute__((ext_vector_type(16))) __bf16 v16bf;
typedef __attribute__((ext_vector_type(8)))  __bf16 v8bf;
typedef __attribute__((ext_vector_type(4)))  __bf16 v4bf;
typedef __attribute__((ext_vector_type(8)))  float  v8f;
typedef __attribute__((ext_vector_type(4)))  unsigned int v4u;
typedef __attribute__((ext_vector_type(8)))  int v8i;
typedef __attribute__((ext_vector_type(4)))  int v4i;

// address-space-qualified pointers for gfx1250 async/TDM builtins
typedef __attribute__((address_space(1))) v4i* gas_v4i;
typedef __attribute__((address_space(3))) v4i* las_v4i;

union BF16x16 { v16bf v; v8bf h[2]; };
union F8      { v8f v; float f[8]; };

__device__ __forceinline__ v8f wmma_bf16(v16bf a, v16bf b, v8f c) {
  // 8 args: (neg_a, A, neg_b, B, c_mod, C, reuse_a, reuse_b)
  return __builtin_amdgcn_wmma_f32_16x16x32_bf16(false, a, false, b, (short)0, c,
                                                 false, false);
}

__device__ __forceinline__ float gelu_f(float v) {
  return 0.5f * v * (1.0f + erff(v * 0.70710678118654752f));
}

// ---------------------------------------------------------------------------
// gfx1250 async / TDM feature detection (compile-safe fallbacks)
// ---------------------------------------------------------------------------
#if defined(__HIP_DEVICE_COMPILE__) && __has_builtin(__builtin_amdgcn_tensor_load_to_lds) && __has_builtin(__builtin_amdgcn_s_wait_tensorcnt)
#define USE_TDM 1
#else
#define USE_TDM 0
#endif

#if defined(__HIP_DEVICE_COMPILE__) && __has_builtin(__builtin_amdgcn_global_load_async_to_lds_b128)
#define USE_ASYNC 1
#else
#define USE_ASYNC 0
#endif

__device__ __forceinline__ unsigned lds_lo(const void* p) {
  // flat shared-memory address: low 32 bits are the LDS byte offset
  return (unsigned)(unsigned long long)p;
}

__device__ __forceinline__ gas_v4i to_global_v4i(const void* p) {
  return (gas_v4i)(unsigned long long)p;          // global VA == flat address
}
__device__ __forceinline__ las_v4i to_lds_v4i(const void* p) {
  return (las_v4i)(unsigned)(unsigned long long)p;  // LDS offset = low 32 bits
}

#if USE_TDM
// Build a D# (groups 0..3 per CDNA5 ISA 8.3-8.6) for a 2D bf16 tile load into
// LDS with row padding (pad_interval=16 DWORDs, pad_amount=4 DWORDs -> 80B
// LDS row pitch for a 64B tile row), and issue TENSOR_LOAD_TO_LDS.
__device__ __forceinline__ void tdm_load_tile_bf16(
    const void* gaddr, unsigned lds_addr, unsigned tensor_w, unsigned tensor_h,
    unsigned tile_w, unsigned tile_h, unsigned row_stride_elems) {
  unsigned long long ga = (unsigned long long)gaddr;
  v4u g0;
  g0[0] = 1u;                                   // count=1 (valid), user mode
  g0[1] = lds_addr;                             // lds_addr  (bits 63:32)
  g0[2] = (unsigned)(ga & 0xffffffffu);         // global_addr[31:0]  (95:64)
  g0[3] = (unsigned)((ga >> 32) & 0x01ffffffu)  // global_addr[56:32] (120:96)
        | (2u << 30);                           // type=2 ("image")
  v8i g1;
  g1[0] = (1 << 16)                             // data_size = 1 -> 2 bytes
        | (1 << 20)                             // pad_enable
        | (3 << 22)                             // pad_interval: 16 DWORDs
        | (3 << 25);                            // pad_amount: 4 DWORDs
  g1[1] = (int)((tensor_w & 0xffffu) << 16);    // tensor_dim0[15:0]  @ 63:48
  g1[2] = (int)(((tensor_w >> 16) & 0xffffu)    // tensor_dim0[31:16] @ 79:64
        | ((tensor_h & 0xffffu) << 16));        // tensor_dim1[15:0]  @ 95:80
  g1[3] = (int)(((tensor_h >> 16) & 0xffffu)    // tensor_dim1[31:16] @ 111:96
        | ((tile_w & 0xffffu) << 16));          // tile_dim0          @ 127:112
  g1[4] = (int)(tile_h & 0xffffu);              // tile_dim1 (tile_dim2 = 0)
  g1[5] = (int)row_stride_elems;                // tensor_dim0_stride[31:0]
  g1[6] = 0;                                    // stride[47:32], dim1_stride lo
  g1[7] = 0;
  v4i g2 = {0, 0, 0, 0};                        // 2D: groups 2/3 unused
  v4i g3 = {0, 0, 0, 0};
  v8i g4 = {0, 0, 0, 0, 0, 0, 0, 0};            // 6-arg form: extra group
  __builtin_amdgcn_tensor_load_to_lds(g0, g1, g2, g3, g4, 0);
}
#endif

#if USE_ASYNC
__device__ __forceinline__ void wait_async0() {
#if __has_builtin(__builtin_amdgcn_s_wait_asynccnt)
  __builtin_amdgcn_s_wait_asynccnt(0);
#else
  asm volatile("s_wait_asynccnt 0x0" ::: "memory");
#endif
}
#endif

// Problem constants
#define BATCH 2
#define DIMC  256
#define HH    64
#define WWID  64
#define NPIX  4096            // H*W
#define HEADS 8
#define HG    2
#define GC    128
#define HC    32
#define RH    32
#define RW    32
#define RTOT  1024
#define NWIN  64
#define WT    64
#define MLPH  1024

// ---------------------------------------------------------------------------
// f32 -> bf16 cast
// ---------------------------------------------------------------------------
__global__ void cast_f32_bf16_kernel(const float* __restrict__ in,
                                     __bf16* __restrict__ out, int n) {
  int i = blockIdx.x * 256 + threadIdx.x;
  if (i < n) out[i] = (__bf16)in[i];
}

// ---------------------------------------------------------------------------
// LN1: x (B,C,H,W) channel-major -> xa pixel-major f32 + bf16 [8192][256]
// ---------------------------------------------------------------------------
__global__ __launch_bounds__(256) void ln1_kernel(const float* __restrict__ x,
    const float* __restrict__ g, const float* __restrict__ bta,
    float* __restrict__ xa_f32, __bf16* __restrict__ xa_bf16) {
  int by = blockIdx.x;             // b*64 + y
  int b = by >> 6, y = by & 63;
  int tid = threadIdx.x;
  int cc = tid >> 6, xx = tid & 63;
  const float* xp = x + (size_t)b * (DIMC * NPIX) + (size_t)y * WWID + xx;
  float s = 0.f, ss = 0.f;
  for (int j = 0; j < 64; j++) {
    float v = xp[(size_t)(cc + 4 * j) * NPIX];
    s += v; ss += v * v;
  }
  __shared__ float sm[4][64], sq[4][64], mean[64], rstd[64];
  sm[cc][xx] = s; sq[cc][xx] = ss;
  __syncthreads();
  if (tid < 64) {
    float S = sm[0][tid] + sm[1][tid] + sm[2][tid] + sm[3][tid];
    float Q = sq[0][tid] + sq[1][tid] + sq[2][tid] + sq[3][tid];
    float mu = S * (1.f / 256.f);
    float var = Q * (1.f / 256.f) - mu * mu;
    mean[tid] = mu; rstd[tid] = rsqrtf(var + 1e-5f);
  }
  __syncthreads();
  float mu = mean[xx], rs = rstd[xx];
  __shared__ float tile[64][4];
  size_t pixbase = (size_t)b * NPIX + (size_t)y * WWID;
  for (int j = 0; j < 64; j++) {
    int c = cc + 4 * j;
    float v = xp[(size_t)c * NPIX];
    float o = (v - mu) * rs * g[c] + bta[c];
    tile[xx][cc] = o;
    __syncthreads();
    int x2 = tid >> 2, c2 = tid & 3;
    float o2 = tile[x2][c2];
    size_t addr = (pixbase + x2) * DIMC + 4 * j + c2;
    xa_f32[addr] = o2;
    xa_bf16[addr] = (__bf16)o2;
    __syncthreads();
  }
}

// ---------------------------------------------------------------------------
// Tiled WMMA GEMM: C[M][N] = A[M][K] * W[N][K]^T  (A,W bf16; acc f32)
// Block 128x64 tile, 256 threads = 8 waves (4x2), each wave 32x32 (2x2 WMMA).
// LDS tiles fed by TDM tensor_load_to_lds when available.
// EPI: 0 f32+bias | 1 bf16+bias | 2 f32+bias+channel-major residual (proj)
//      3 bf16+bias+gelu (mlp1)  | 4 f32+bias+row-major residual (mlp2)
// ---------------------------------------------------------------------------
template <int EPI>
__global__ __launch_bounds__(256) void gemm_bf16_kernel(
    const __bf16* __restrict__ A, const __bf16* __restrict__ W,
    const float* __restrict__ bias, const float* __restrict__ resid,
    void* __restrict__ Cout, int M, int N, int K) {
  __shared__ __align__(16) __bf16 As[128][40];
  __shared__ __align__(16) __bf16 Bs[64][40];
  int tid = threadIdx.x;
  int wid = tid >> 5, lane = tid & 31, half = lane >> 4, ln = lane & 15;
  int wm = wid & 3, wn = wid >> 2;
  int m0 = blockIdx.y * 128, n0 = blockIdx.x * 64;
  F8 acc[2][2];
  for (int i = 0; i < 2; i++)
    for (int j = 0; j < 2; j++)
      for (int gg = 0; gg < 8; gg++) acc[i][j].f[gg] = 0.f;

  for (int k0 = 0; k0 < K; k0 += 32) {
    __syncthreads();
#if USE_TDM
    if (wid == 0) {   // one wave drives the Tensor Data Mover for both tiles
      tdm_load_tile_bf16(&A[(size_t)m0 * K + k0], lds_lo(&As[0][0]),
                         (unsigned)K, (unsigned)M, 32u, 128u, (unsigned)K);
      tdm_load_tile_bf16(&W[(size_t)n0 * K + k0], lds_lo(&Bs[0][0]),
                         (unsigned)K, (unsigned)N, 32u, 64u, (unsigned)K);
      __builtin_amdgcn_s_wait_tensorcnt((short)0);
    }
#else
    // A tile 128x32
    for (int i = tid; i < 512; i += 256) {
      int r = i >> 2, cp = i & 3;
      *(v8bf*)&As[r][cp * 8] =
          *(const v8bf*)&A[(size_t)(m0 + r) * K + k0 + cp * 8];
    }
    // W tile 64x32 (B[k][n] = W[n][k], kept row-major in K)
    {
      int r = tid >> 2, cp = tid & 3;
      *(v8bf*)&Bs[r][cp * 8] =
          *(const v8bf*)&W[(size_t)(n0 + r) * K + k0 + cp * 8];
    }
    if (k0 + 32 < K)  // gfx1250 global_prefetch_b8
      __builtin_prefetch(
          (const void*)&A[(size_t)(m0 + (tid >> 1)) * K + k0 + 32 + (tid & 1) * 16],
          0, 0);
#endif
    __syncthreads();
    BF16x16 af[2], bf[2];
    for (int i = 0; i < 2; i++) {
      int row = wm * 32 + i * 16 + ln;
      af[i].h[0] = *(const v8bf*)&As[row][half * 8];
      af[i].h[1] = *(const v8bf*)&As[row][16 + half * 8];
    }
    for (int j = 0; j < 2; j++) {
      int col = wn * 32 + j * 16 + ln;
      bf[j].h[0] = *(const v8bf*)&Bs[col][half * 8];
      bf[j].h[1] = *(const v8bf*)&Bs[col][16 + half * 8];
    }
    for (int i = 0; i < 2; i++)
      for (int j = 0; j < 2; j++)
        acc[i][j].v = wmma_bf16(af[i].v, bf[j].v, acc[i][j].v);
  }
  // epilogue (D layout: VGPR g -> row g + 8*half, col = lane%16)
  for (int i = 0; i < 2; i++)
    for (int j = 0; j < 2; j++)
      for (int gg = 0; gg < 8; gg++) {
        int row = m0 + wm * 32 + i * 16 + gg + 8 * half;
        int col = n0 + wn * 32 + j * 16 + ln;
        float v = acc[i][j].f[gg] + bias[col];
        if (EPI == 2)  // residual from (B,C,H,W): H*W = 4096
          v += resid[((size_t)(row >> 12) * DIMC + col) * NPIX + (row & 4095)];
        if (EPI == 4) v += resid[(size_t)row * N + col];
        if (EPI == 3) v = gelu_f(v);
        if (EPI == 0 || EPI == 2 || EPI == 4)
          ((float*)Cout)[(size_t)row * N + col] = v;
        else
          ((__bf16*)Cout)[(size_t)row * N + col] = (__bf16)v;
      }
}

// ---------------------------------------------------------------------------
// Depthwise 3x3 stride-2 pad-1 conv (feature_group_count=GC), optional BN+gelu
// ---------------------------------------------------------------------------
template <bool PIXIN, bool DOBN>
__global__ __launch_bounds__(256) void dwconv_kernel(
    const float* __restrict__ in, const float* __restrict__ wgt,
    float* __restrict__ out, int Hi, int Ho,
    const float* __restrict__ bn_g, const float* __restrict__ bn_b,
    const float* __restrict__ bn_m, const float* __restrict__ bn_v) {
  int idx = blockIdx.x * 256 + threadIdx.x;
  int total = 4 * GC * Ho * Ho;
  if (idx >= total) return;
  int p = idx % (Ho * Ho);
  int tmp = idx / (Ho * Ho);
  int c = tmp % GC, bg = tmp / GC;
  int oy = p / Ho, ox = p % Ho;
  float s = 0.f;
  for (int ky = 0; ky < 3; ky++) {
    int iy = oy * 2 - 1 + ky;
    if (iy < 0 || iy >= Hi) continue;
    for (int kx = 0; kx < 3; kx++) {
      int ix = ox * 2 - 1 + kx;
      if (ix < 0 || ix >= Hi) continue;
      float v;
      if (PIXIN) {
        int b = bg >> 1;
        v = in[((size_t)b * NPIX + iy * WWID + ix) * DIMC + (bg & 1) * GC + c];
      } else {
        v = in[(((size_t)bg * GC + c) * Hi + iy) * Hi + ix];
      }
      s += v * wgt[c * 9 + ky * 3 + kx];
    }
  }
  if (DOBN) {
    s = (s - bn_m[c]) * rsqrtf(bn_v[c] + 1e-5f) * bn_g[c] + bn_b[c];
    s = gelu_f(s);
  }
  out[(((size_t)bg * GC + c) * Ho + oy) * Ho + ox] = s;
}

// ---------------------------------------------------------------------------
// offset head: off = off4_w @ o ; grid = tanh(off)*ranges*FACTOR + ref
// ---------------------------------------------------------------------------
__global__ __launch_bounds__(256) void offset_grid_kernel(
    const float* __restrict__ t3, const float* __restrict__ w4,
    float* __restrict__ grid) {
  int id = blockIdx.x * 256 + threadIdx.x;   // (bg*64+nw)*1024 + r
  if (id >= 4 * NWIN * RTOT) return;
  int r = id & 1023;
  int bgnw = id >> 10;
  int nw = bgnw & 63, bg = bgnw >> 6;
  const float* tp = t3 + (size_t)bg * (GC * 64) + nw;   // + c*64
  const float* w0 = w4 + (size_t)r * GC;
  const float* w1 = w4 + (size_t)(RTOT + r) * GC;
  float d0 = 0.f, d1 = 0.f;
  for (int c = 0; c < GC; c++) {
    float v = tp[(size_t)c * 64];
    d0 += v * w0[c];
    d1 += v * w1[c];
  }
  float ry = (float)(r >> 5), rx = (float)(r & 31);
  float g0 = tanhf(d0) * (2.0f / 64.0f) + (ry * 4.0f / 63.0f - 1.0f);
  float g1 = tanhf(d1) * (2.0f / 64.0f) + (rx * 4.0f / 63.0f - 1.0f);
  grid[(size_t)id * 2 + 0] = g0;
  grid[(size_t)id * 2 + 1] = g1;
}

// ---------------------------------------------------------------------------
// Bilinear grid-sample: one wave per sample point, lane covers 4 channels.
// ---------------------------------------------------------------------------
__global__ __launch_bounds__(256) void grid_sample_kernel(
    const float* __restrict__ xa, const float* __restrict__ grid,
    __bf16* __restrict__ samp) {
  int gw = blockIdx.x * 8 + (threadIdx.x >> 5);
  int lane = threadIdx.x & 31;
  int r = gw & 1023;
  int bgnw = gw >> 10;
  int nw = bgnw & 63, bg = bgnw >> 6;
  int b = bg >> 1, hg = bg & 1;
  float rows = (grid[(size_t)gw * 2 + 0] + 1.0f) * 0.5f * 63.0f;
  float cols = (grid[(size_t)gw * 2 + 1] + 1.0f) * 0.5f * 63.0f;
  float r0f = floorf(rows), c0f = floorf(cols);
  float wr = rows - r0f, wc = cols - c0f;
  int r0 = (int)r0f, c0 = (int)c0f;
  float acc0 = 0.f, acc1 = 0.f, acc2 = 0.f, acc3 = 0.f;
  float wts[4] = {(1.f - wr) * (1.f - wc), (1.f - wr) * wc,
                  wr * (1.f - wc), wr * wc};
  int rr[4] = {r0, r0, r0 + 1, r0 + 1};
  int cc[4] = {c0, c0 + 1, c0, c0 + 1};
  for (int k = 0; k < 4; k++) {
    int ri = rr[k], ci = cc[k];
    if (ri < 0 || ri > 63 || ci < 0 || ci > 63) continue;
    const float4 v = *(const float4*)&xa[((size_t)b * NPIX + ri * WWID + ci) * DIMC +
                                         hg * GC + lane * 4];
    float w = wts[k];
    acc0 += w * v.x; acc1 += w * v.y; acc2 += w * v.z; acc3 += w * v.w;
  }
  v4bf o;
  o[0] = (__bf16)acc0; o[1] = (__bf16)acc1;
  o[2] = (__bf16)acc2; o[3] = (__bf16)acc3;
  *(v4bf*)&samp[((size_t)(b * NWIN + nw) * RTOT + r) * DIMC + hg * GC + lane * 4] = o;
}

// ---------------------------------------------------------------------------
// Flash attention per (n = b*8+head, window w): 128 threads = 4 waves,
// wave owns 16 query rows; R streamed in chunks of 64; WMMA for QK^T and PV.
// K chunks fetched with gfx1250 async global->LDS loads when available.
// ---------------------------------------------------------------------------
__global__ __launch_bounds__(128) void attn_kernel(
    const float* __restrict__ qpix, const __bf16* __restrict__ ktmp,
    const __bf16* __restrict__ vtmp, const float* __restrict__ posembed,
    __bf16* __restrict__ aout) {
  __shared__ __align__(16) __bf16 Qs[64][40];
  __shared__ __align__(16) __bf16 Ks[64][40];       // [r_local][c]
  __shared__ __align__(16) __bf16 Vs[32][72];       // [c][r_local]
  __shared__ __align__(16) __bf16 Ps[4][16][72];    // per-wave P staging
  int blk = blockIdx.x;
  int n = blk >> 6, w = blk & 63;
  int b = n >> 3, head = n & 7;
  int wy = w >> 3, wx = w & 7;
  int tid = threadIdx.x, wv = tid >> 5, lane = tid & 31;
  int half = lane >> 4, ln = lane & 15;
  const float scale = 0.0625f;  // 256^-0.5

  for (int i = tid; i < 64 * 32; i += 128) {
    int qr = i >> 5, c = i & 31;
    int y = wy * 8 + (qr >> 3), x = wx * 8 + (qr & 7);
    Qs[qr][c] =
        (__bf16)(qpix[((size_t)b * NPIX + y * WWID + x) * DIMC + head * HC + c] * scale);
  }
  __syncthreads();
  BF16x16 aq;
  aq.h[0] = *(const v8bf*)&Qs[wv * 16 + ln][half * 8];
  aq.h[1] = *(const v8bf*)&Qs[wv * 16 + ln][16 + half * 8];

  float mrow[8], lrow[8];
  F8 oacc[2];
  for (int g = 0; g < 8; g++) { mrow[g] = -3.0e38f; lrow[g] = 0.f; }
  for (int ct = 0; ct < 2; ct++)
    for (int g = 0; g < 8; g++) oacc[ct].f[g] = 0.f;

  size_t rowbase = (size_t)(b * NWIN + w) * RTOT;
  for (int rc = 0; rc < 16; rc++) {
    __syncthreads();
    int rb = rc * 64;
#if USE_ASYNC
    for (int i = tid; i < 256; i += 128) {        // K chunk via ASYNCcnt path
      int rl = i >> 2, cp = i & 3;
      __builtin_amdgcn_global_load_async_to_lds_b128(
          to_global_v4i(&ktmp[(rowbase + rb + rl) * DIMC + head * HC + cp * 8]),
          to_lds_v4i(&Ks[rl][cp * 8]), 0, 0);
    }
#else
    for (int i = tid; i < 256; i += 128) {        // K chunk: 64 r x 32 c
      int rl = i >> 2, cp = i & 3;
      *(v8bf*)&Ks[rl][cp * 8] =
          *(const v8bf*)&ktmp[(rowbase + rb + rl) * DIMC + head * HC + cp * 8];
    }
#endif
    for (int i = tid; i < 512; i += 128) {        // V chunk transposed to [c][r]
      int rl = i >> 3, cp = i & 7;
      const __bf16* src = &vtmp[(rowbase + rb + rl) * DIMC + head * HC + cp * 4];
      Vs[cp * 4 + 0][rl] = src[0];
      Vs[cp * 4 + 1][rl] = src[1];
      Vs[cp * 4 + 2][rl] = src[2];
      Vs[cp * 4 + 3][rl] = src[3];
    }
#if USE_ASYNC
    wait_async0();
#endif
    __syncthreads();

    F8 S[4];
    for (int nt = 0; nt < 4; nt++) {
      BF16x16 bk;
      bk.h[0] = *(const v8bf*)&Ks[nt * 16 + ln][half * 8];
      bk.h[1] = *(const v8bf*)&Ks[nt * 16 + ln][16 + half * 8];
      v8f z = {};
      S[nt].v = wmma_bf16(aq.v, bk.v, z);
    }
    // relative-position bias gather
    for (int nt = 0; nt < 4; nt++)
      for (int g = 0; g < 8; g++) {
        int qr = wv * 16 + g + 8 * half;
        int y = wy * 8 + (qr >> 3), x = wx * 8 + (qr & 7);
        int rg = rb + nt * 16 + ln;
        int ry = rg >> 5, rx = rg & 31;
        S[nt].f[g] += posembed[(size_t)head * 16129 +
                               (ry * 2 - y + 63) * 127 + (rx * 2 - x + 63)];
      }
    // online softmax (rows split across 16-lane halves)
    float mnew[8], alpha[8];
    for (int g = 0; g < 8; g++) {
      float mt = fmaxf(fmaxf(S[0].f[g], S[1].f[g]), fmaxf(S[2].f[g], S[3].f[g]));
      mt = fmaxf(mt, __shfl_xor(mt, 1));
      mt = fmaxf(mt, __shfl_xor(mt, 2));
      mt = fmaxf(mt, __shfl_xor(mt, 4));
      mt = fmaxf(mt, __shfl_xor(mt, 8));
      mnew[g] = fmaxf(mrow[g], mt);
      alpha[g] = __expf(mrow[g] - mnew[g]);
      mrow[g] = mnew[g];
    }
    for (int nt = 0; nt < 4; nt++)
      for (int g = 0; g < 8; g++) S[nt].f[g] = __expf(S[nt].f[g] - mnew[g]);
    for (int g = 0; g < 8; g++) {
      float ps = S[0].f[g] + S[1].f[g] + S[2].f[g] + S[3].f[g];
      ps += __shfl_xor(ps, 1);
      ps += __shfl_xor(ps, 2);
      ps += __shfl_xor(ps, 4);
      ps += __shfl_xor(ps, 8);
      lrow[g] = lrow[g] * alpha[g] + ps;
    }
    for (int ct = 0; ct < 2; ct++)
      for (int g = 0; g < 8; g++) oacc[ct].f[g] *= alpha[g];
    // stage P (D-layout -> A-layout via per-wave LDS)
    for (int nt = 0; nt < 4; nt++)
      for (int g = 0; g < 8; g++)
        Ps[wv][g + 8 * half][nt * 16 + ln] = (__bf16)S[nt].f[g];
    for (int kh = 0; kh < 2; kh++) {
      BF16x16 ap;
      ap.h[0] = *(const v8bf*)&Ps[wv][ln][kh * 32 + half * 8];
      ap.h[1] = *(const v8bf*)&Ps[wv][ln][kh * 32 + 16 + half * 8];
      for (int ct = 0; ct < 2; ct++) {
        BF16x16 bv;
        bv.h[0] = *(const v8bf*)&Vs[ct * 16 + ln][kh * 32 + half * 8];
        bv.h[1] = *(const v8bf*)&Vs[ct * 16 + ln][kh * 32 + 16 + half * 8];
        oacc[ct].v = wmma_bf16(ap.v, bv.v, oacc[ct].v);
      }
    }
  }
  // write attention output, pixel-major bf16 [8192][256]
  for (int ct = 0; ct < 2; ct++)
    for (int g = 0; g < 8; g++) {
      int qr = wv * 16 + g + 8 * half;
      int y = wy * 8 + (qr >> 3), x = wx * 8 + (qr & 7);
      float val = oacc[ct].f[g] / lrow[g];
      aout[((size_t)b * NPIX + y * WWID + x) * DIMC + head * HC + ct * 16 + ln] =
          (__bf16)val;
    }
}

// ---------------------------------------------------------------------------
// LN2 on pixel-major input -> bf16 pixel-major. One wave per pixel.
// ---------------------------------------------------------------------------
__global__ __launch_bounds__(256) void ln2_kernel(const float* __restrict__ xin,
    const float* __restrict__ g, const float* __restrict__ bta,
    __bf16* __restrict__ out) {
  int wv = threadIdx.x >> 5, lane = threadIdx.x & 31;
  size_t pix = (size_t)blockIdx.x * 8 + wv;
  const float* p = xin + pix * DIMC;
  float vals[8], s = 0.f, ss = 0.f;
  for (int i = 0; i < 8; i++) {
    float v = p[i * 32 + lane];
    vals[i] = v; s += v; ss += v * v;
  }
  for (int m = 1; m < 32; m <<= 1) { s += __shfl_xor(s, m); ss += __shfl_xor(ss, m); }
  float mu = s * (1.f / 256.f);
  float var = ss * (1.f / 256.f) - mu * mu;
  float rs = rsqrtf(var + 1e-5f);
  for (int i = 0; i < 8; i++) {
    int c = i * 32 + lane;
    out[pix * DIMC + c] = (__bf16)((vals[i] - mu) * rs * g[c] + bta[c]);
  }
}

// ---------------------------------------------------------------------------
// Final: pixel-major [B*4096][256] f32 -> d_out (B,256,64,64), LDS tiled
// ---------------------------------------------------------------------------
__global__ __launch_bounds__(256) void final_transpose_kernel(
    const float* __restrict__ xin, float* __restrict__ out) {
  __shared__ float tile[32][33];
  int blk = blockIdx.x;            // b * (8 ctiles) * (128 ptiles)
  int b = blk / 1024;
  int rem = blk % 1024;
  int ct = rem / 128, pt = rem % 128;
  int c0 = ct * 32, p0 = pt * 32;
  int tid = threadIdx.x;
  int cc = tid & 31, pr = tid >> 5;
  for (int it = 0; it < 4; it++) {
    int p = pr + it * 8;
    tile[p][cc] = xin[((size_t)b * NPIX + p0 + p) * DIMC + c0 + cc];
  }
  __syncthreads();
  int pr2 = tid & 31, cc2 = tid >> 5;
  for (int it = 0; it < 4; it++) {
    int c = cc2 + it * 8;
    out[((size_t)b * DIMC + c0 + c) * NPIX + p0 + pr2] = tile[pr2][c];
  }
}

// ---------------------------------------------------------------------------
// Host launcher
// ---------------------------------------------------------------------------
extern "C" void kernel_launch(void* const* d_in, const int* in_sizes, int n_in,
                              void* d_out, int out_size, void* d_ws, size_t ws_size,
                              hipStream_t stream) {
  const float* x       = (const float*)d_in[0];
  const float* ln1_g   = (const float*)d_in[1];
  const float* ln1_b   = (const float*)d_in[2];
  const float* q_w     = (const float*)d_in[3];
  const float* q_b     = (const float*)d_in[4];
  const float* k_w     = (const float*)d_in[5];
  const float* k_b     = (const float*)d_in[6];
  const float* v_w     = (const float*)d_in[7];
  const float* v_b     = (const float*)d_in[8];
  const float* off1_w  = (const float*)d_in[9];
  const float* off2_w  = (const float*)d_in[10];
  const float* off3_w  = (const float*)d_in[11];
  const float* bn_g    = (const float*)d_in[12];
  const float* bn_b    = (const float*)d_in[13];
  const float* bn_m    = (const float*)d_in[14];
  const float* bn_v    = (const float*)d_in[15];
  const float* off4_w  = (const float*)d_in[16];
  const float* posembed= (const float*)d_in[17];
  const float* proj_w  = (const float*)d_in[18];
  const float* proj_b  = (const float*)d_in[19];
  const float* ln2_g   = (const float*)d_in[20];
  const float* ln2_b   = (const float*)d_in[21];
  const float* mlp_w1  = (const float*)d_in[22];
  const float* mlp_b1  = (const float*)d_in[23];
  const float* mlp_w2  = (const float*)d_in[24];
  const float* mlp_b2  = (const float*)d_in[25];
  (void)in_sizes; (void)n_in; (void)out_size; (void)ws_size;

  char* ws = (char*)d_ws;
  size_t off = 0;
  auto take = [&](size_t bytes) -> void* {
    void* p = ws + off;
    off += (bytes + 255) & ~(size_t)255;
    return p;
  };
  const size_t NPX = (size_t)BATCH * NPIX;            // 8192 pixels
  const size_t MSAMP = (size_t)BATCH * NWIN * RTOT;   // 131072 rows

  __bf16* qw_bf   = (__bf16*)take(65536 * 2);
  __bf16* kw_bf   = (__bf16*)take(65536 * 2);
  __bf16* vw_bf   = (__bf16*)take(65536 * 2);
  __bf16* pw_bf   = (__bf16*)take(65536 * 2);
  __bf16* w1_bf   = (__bf16*)take(262144 * 2);
  __bf16* w2_bf   = (__bf16*)take(262144 * 2);
  float*  xa_f32  = (float*) take(NPX * DIMC * 4);
  __bf16* xa_bf   = (__bf16*)take(NPX * DIMC * 2);
  float*  q_pix   = (float*) take(NPX * DIMC * 4);
  float*  t1      = (float*) take((size_t)4 * GC * 32 * 32 * 4);
  float*  t2      = (float*) take((size_t)4 * GC * 16 * 16 * 4);
  float*  t3      = (float*) take((size_t)4 * GC * 8 * 8 * 4);
  float*  gridb   = (float*) take((size_t)4 * NWIN * RTOT * 2 * 4);
  __bf16* samp    = (__bf16*)take(MSAMP * DIMC * 2);
  __bf16* ktmp    = (__bf16*)take(MSAMP * DIMC * 2);
  __bf16* vtmp    = (__bf16*)take(MSAMP * DIMC * 2);
  __bf16* aoutb   = (__bf16*)take(NPX * DIMC * 2);
  float*  x2_pix  = (float*) take(NPX * DIMC * 4);
  __bf16* xm_bf   = (__bf16*)take(NPX * DIMC * 2);
  __bf16* hdn     = (__bf16*)take(NPX * MLPH * 2);
  float*  x3_pix  = (float*) take(NPX * DIMC * 4);

  // 1) weight casts
  cast_f32_bf16_kernel<<<256, 256, 0, stream>>>(q_w, qw_bf, 65536);
  cast_f32_bf16_kernel<<<256, 256, 0, stream>>>(k_w, kw_bf, 65536);
  cast_f32_bf16_kernel<<<256, 256, 0, stream>>>(v_w, vw_bf, 65536);
  cast_f32_bf16_kernel<<<256, 256, 0, stream>>>(proj_w, pw_bf, 65536);
  cast_f32_bf16_kernel<<<1024, 256, 0, stream>>>(mlp_w1, w1_bf, 262144);
  cast_f32_bf16_kernel<<<1024, 256, 0, stream>>>(mlp_w2, w2_bf, 262144);

  // 2) LN1
  ln1_kernel<<<BATCH * HH, 256, 0, stream>>>(x, ln1_g, ln1_b, xa_f32, xa_bf);

  // 3) q = xa @ q_w^T + q_b  (f32 out, pixel-major)
  gemm_bf16_kernel<0><<<dim3(4, 64), 256, 0, stream>>>(
      xa_bf, qw_bf, q_b, nullptr, q_pix, (int)NPX, DIMC, DIMC);

  // 4-6) depthwise conv chain + BN + gelu
  dwconv_kernel<true, false><<<2048, 256, 0, stream>>>(
      q_pix, off1_w, t1, 64, 32, nullptr, nullptr, nullptr, nullptr);
  dwconv_kernel<false, false><<<512, 256, 0, stream>>>(
      t1, off2_w, t2, 32, 16, nullptr, nullptr, nullptr, nullptr);
  dwconv_kernel<false, true><<<128, 256, 0, stream>>>(
      t2, off3_w, t3, 16, 8, bn_g, bn_b, bn_m, bn_v);

  // 7) offset head -> sampling grid
  offset_grid_kernel<<<1024, 256, 0, stream>>>(t3, off4_w, gridb);

  // 8) bilinear grid-sample -> samp bf16
  grid_sample_kernel<<<32768, 256, 0, stream>>>(xa_f32, gridb, samp);

  // 9-10) k, v projections over sampled rows (bf16 out)
  gemm_bf16_kernel<1><<<dim3(4, 1024), 256, 0, stream>>>(
      samp, kw_bf, k_b, nullptr, ktmp, (int)MSAMP, DIMC, DIMC);
  gemm_bf16_kernel<1><<<dim3(4, 1024), 256, 0, stream>>>(
      samp, vw_bf, v_b, nullptr, vtmp, (int)MSAMP, DIMC, DIMC);

  // 11) flash attention with rel-pos bias
  attn_kernel<<<1024, 128, 0, stream>>>(q_pix, ktmp, vtmp, posembed, aoutb);

  // 12) proj + residual(x, channel-major) -> x2 pixel-major
  gemm_bf16_kernel<2><<<dim3(4, 64), 256, 0, stream>>>(
      aoutb, pw_bf, proj_b, x, x2_pix, (int)NPX, DIMC, DIMC);

  // 13) LN2 -> bf16
  ln2_kernel<<<1024, 256, 0, stream>>>(x2_pix, ln2_g, ln2_b, xm_bf);

  // 14) MLP1 + gelu (bf16 out)
  gemm_bf16_kernel<3><<<dim3(16, 64), 256, 0, stream>>>(
      xm_bf, w1_bf, mlp_b1, nullptr, hdn, (int)NPX, MLPH, DIMC);

  // 15) MLP2 + residual(x2 row-major) -> x3 pixel-major
  gemm_bf16_kernel<4><<<dim3(4, 64), 256, 0, stream>>>(
      hdn, w2_bf, mlp_b2, x2_pix, x3_pix, (int)NPX, DIMC, MLPH);

  // 16) transpose to (B, C, H, W) output
  final_transpose_kernel<<<2048, 256, 0, stream>>>(x3_pix, (float*)d_out);
}
